// WrappedSeqback_70935679861224
// MI455X (gfx1250) — matrix operand order repaired and checked by reading, not compiled, not run on hardware
//
#include <hip/hip_runtime.h>
#include <hip/hip_bf16.h>

// ---------------------------------------------------------------------------
// WrappedSeqback forward for MI455X (gfx1250, wave32, WMMA bf16 path).
// Sequential 128-step LSTM decode; per-step [16x512]x[512x32000] argmax GEMM
// runs on V_WMMA_F32_16X16X32_BF16 with fp32 accumulation. All bf16 weights
// (~42 MB) live in the 192 MB L2 across steps. A-operand staging into LDS
// uses the Tensor Data Mover (tensor_load_to_lds + s_wait_tensorcnt).
// ---------------------------------------------------------------------------

typedef __bf16 bf16_t;
typedef __attribute__((ext_vector_type(16))) __bf16 v16bf;
typedef __attribute__((ext_vector_type(8)))  float  v8f;
typedef unsigned int u32x4 __attribute__((ext_vector_type(4)));
typedef int i32x4 __attribute__((ext_vector_type(4)));
typedef int i32x8 __attribute__((ext_vector_type(8)));

#define BB   16      // batch
#define SS   128     // decode steps
#define TT   160     // sentence length
#define HH   512     // hidden
#define EE   512     // embed
#define RDD  128     // relation embed
#define VV   32000   // vocab
#define G4H  2048    // 4*H
#define KIN  (RDD + EE)   // 640, layer-0 input width

union FragU { v16bf v; uint4 q[2]; };

// A-matrix fragment (16x32 bf16 tile), contiguous k order (B uses the same):
// lane m (0-15): elems 0-15 = K k0+0..15
// lane m+16    : elems 0-15 = K k0+16..31
__device__ __forceinline__ v16bf load_frag_a(const bf16_t* base, int stride,
                                             int m, int khalf, int k0) {
  FragU u;
  const bf16_t* p0 = base + (size_t)m * stride + k0 + khalf * 16;
  u.q[0] = *(const uint4*)p0;
  u.q[1] = *(const uint4*)(p0 + 8);
  return u.v;
}

// B-matrix fragment (32x16 bf16 tile) from row-major [N,K] weights:
// lane l: n = l&15, khalf = l>>4; 16 contiguous bf16 at W[ncb+n][k0+khalf*16].
__device__ __forceinline__ v16bf load_frag_b(const bf16_t* W, int stride,
                                             int ncb, int lane, int k0) {
  const int n = lane & 15, khalf = lane >> 4;
  FragU u;
  const bf16_t* p = W + (size_t)(ncb + n) * stride + k0 + khalf * 16;
  u.q[0] = *(const uint4*)p;
  u.q[1] = *(const uint4*)(p + 8);
  return u.v;
}

__device__ __forceinline__ float sigmoidf_(float x) {
  return 1.0f / (1.0f + __expf(-x));
}

// ---------------------- Tensor Data Mover: global -> LDS --------------------
// Flat 1-D copy of `ndwords` dwords (D# per CDNA5 ISA 8.3/8.4):
//   group0: count=1 | lds_addr | global_addr[56:0] | type=2
//   group1: data_size=4B, tensor_dim0=tile_dim0=ndwords, tensor_dim1=1,
//           tile_dim1/2 unused, tensor_dim0_stride=ndwords
// Issue from one wave, wait TENSORcnt, then barrier publishes LDS.
__device__ __forceinline__ void tdm_load_to_lds(void* lds_dst, const void* gsrc,
                                                unsigned ndwords) {
  // generic LDS address: low 32 bits are the LDS byte offset (flat aperture)
  const unsigned lds_off = (unsigned)(uintptr_t)lds_dst;
  const unsigned long long ga = (unsigned long long)(uintptr_t)gsrc;
  u32x4 g0;
  g0[0] = 1u;                                              // count=1, user mode
  g0[1] = lds_off;                                         // lds_addr
  g0[2] = (unsigned)ga;                                    // global_addr[31:0]
  g0[3] = (unsigned)((ga >> 32) & 0x01FFFFFFu) | (2u << 30); // addr[56:32]|type=2
  i32x8 g1;
  g1[0] = (int)(2u << 16);                  // wg_mask=0, data_size=2 (4 bytes)
  g1[1] = (int)((ndwords & 0xFFFFu) << 16); // tensor_dim0[15:0] in bits 63:48
  g1[2] = (int)(((ndwords >> 16) & 0xFFFFu) | (1u << 16)); // dim0 hi | tensor_dim1=1
  g1[3] = (int)((ndwords & 0xFFFFu) << 16); // tensor_dim1 hi=0 | tile_dim0
  g1[4] = 0;                                // tile_dim1=0 (unused), tile_dim2=0
  g1[5] = (int)ndwords;                     // tensor_dim0_stride[31:0]
  g1[6] = 0;                                // stride0 hi | stride1 lo
  g1[7] = 0;                                // stride1 hi
  i32x4 z4 = {};
#if defined(__clang_major__) && (__clang_major__ >= 23)
  i32x8 z8 = {};
  __builtin_amdgcn_tensor_load_to_lds(g0, g1, z4, z4, z8, 0);
#else
  __builtin_amdgcn_tensor_load_to_lds(g0, g1, z4, z4, 0);
#endif
}

// --------------------------- one-time conversions ---------------------------
__global__ void f32_to_bf16_kernel(const float* __restrict__ src,
                                   bf16_t* __restrict__ dst, int n) {
  for (int i = blockIdx.x * blockDim.x + threadIdx.x; i < n;
       i += gridDim.x * blockDim.x)
    dst[i] = (bf16_t)src[i];
}

__global__ void init_state_kernel(const float* __restrict__ hiddens,
                                  bf16_t* __restrict__ h0, bf16_t* __restrict__ h1,
                                  float* __restrict__ c0, float* __restrict__ c1) {
  int idx = blockIdx.x * blockDim.x + threadIdx.x;
  if (idx < BB * HH) {
    float v = hiddens[idx];
    h0[idx] = (bf16_t)v;
    h1[idx] = (bf16_t)v;
    c0[idx] = v;
    c1[idx] = v;
  }
}

// ------------------------------ per-step prep -------------------------------
// Decodes previous step's packed argmax, resets accumulator, builds
// x = [rel_emb[rel] | word_embed] as bf16 [16,640].
__global__ void prep_kernel(int s,
                            const int* __restrict__ relations,
                            const int* __restrict__ gold,
                            const int* __restrict__ tmask,
                            const float* __restrict__ embed,
                            const float* __restrict__ rel_emb,
                            unsigned long long* __restrict__ packed,
                            int* __restrict__ res,
                            bf16_t* __restrict__ x) {
  __shared__ int tok_s[BB];
  const int tid = threadIdx.x;
  if (tid < BB) {
    int prev = 0;
    if (s > 0) {
      unsigned long long p = packed[tid];
      prev = 0x7FFFFFFF - (int)(unsigned)(p & 0xFFFFFFFFull);
      res[(s - 1) * BB + tid] = prev;
    }
    packed[tid] = 0ull;  // reset for this step's atomicMax race
    int tok;
    if (s == 0)                tok = 1;                       // SOS / null token
    else if (tmask[tid * SS + s]) tok = gold[tid * SS + (s - 1)];
    else                       tok = prev;                    // model argmax
    tok_s[tid] = tok;
  }
  __syncthreads();
  for (int idx = tid; idx < BB * KIN; idx += blockDim.x) {
    const int b = idx / KIN, e = idx % KIN;
    float v;
    if (e < RDD) v = rel_emb[relations[b * SS + s] * RDD + e];
    else         v = embed[(size_t)tok_s[b] * EE + (e - RDD)];
    x[idx] = (bf16_t)v;
  }
}

// ------------------------------- LSTM layer ---------------------------------
// g = A1 @ W1^T + A2 @ W2^T + bias  (M=16, N=2048), gates -> h', c'.
// One 1024-thread workgroup: 32 waves x 4 N-tiles. Dynamic LDS: A1|A2|g.
// A operands staged into LDS by the Tensor Data Mover.
__global__ __launch_bounds__(1024)
void lstm_layer_kernel(const bf16_t* __restrict__ A1, int K1,
                       const bf16_t* __restrict__ W1,
                       const bf16_t* __restrict__ A2, int K2,
                       const bf16_t* __restrict__ W2,
                       const float* __restrict__ bias,
                       const float* __restrict__ c_in,
                       float* __restrict__ c_out,
                       bf16_t* __restrict__ h_out) {
  extern __shared__ char smem[];
  bf16_t* sA1 = (bf16_t*)smem;
  bf16_t* sA2 = sA1 + BB * K1;
  float*  sG  = (float*)(smem + (size_t)BB * (K1 + K2) * sizeof(bf16_t));

  const int tid = threadIdx.x;
  if (tid == 0) {
    tdm_load_to_lds(sA1, A1, (unsigned)(BB * K1 * sizeof(bf16_t) / 4));
    tdm_load_to_lds(sA2, A2, (unsigned)(BB * K2 * sizeof(bf16_t) / 4));
    __builtin_amdgcn_s_wait_tensorcnt(0);
  }
  __syncthreads();

  const int wave = tid >> 5, lane = tid & 31;
  const int m = lane & 15, khalf = lane >> 4;
  for (int i = 0; i < 4; ++i) {
    const int ncb = (wave * 4 + i) * 16;
    __builtin_prefetch(W1 + (size_t)(ncb + m) * K1, 0, 0);
    v8f acc = {};
    for (int kt = 0; kt < K1; kt += 32) {
      v16bf a = load_frag_a(sA1, K1, m, khalf, kt);
      v16bf b = load_frag_b(W1, K1, ncb, lane, kt);
      acc = __builtin_amdgcn_wmma_f32_16x16x32_bf16(false, a, false, b,
                                                    (short)0, acc, false, false);
    }
    for (int kt = 0; kt < K2; kt += 32) {
      v16bf a = load_frag_a(sA2, K2, m, khalf, kt);
      v16bf b = load_frag_b(W2, K2, ncb, lane, kt);
      acc = __builtin_amdgcn_wmma_f32_16x16x32_bf16(false, a, false, b,
                                                    (short)0, acc, false, false);
    }
    const int n = ncb + m;
    const float bv = bias[n];
    const int mbase = khalf * 8;
#pragma unroll
    for (int r = 0; r < 8; ++r)
      sG[(mbase + r) * G4H + n] = acc[r] + bv;
  }
  __syncthreads();

  for (int idx = tid; idx < BB * HH; idx += blockDim.x) {
    const int b = idx >> 9, j = idx & 511;
    const float gi = sG[b * G4H + j];
    const float gf = sG[b * G4H + 512 + j];
    const float gg = sG[b * G4H + 1024 + j];
    const float go = sG[b * G4H + 1536 + j];
    const float c2 = sigmoidf_(gf) * c_in[idx] + sigmoidf_(gi) * tanhf(gg);
    const float h2 = sigmoidf_(go) * tanhf(c2);
    c_out[idx] = c2;
    h_out[idx] = (bf16_t)h2;
  }
}

// -------------------------- logits GEMM + argmax ----------------------------
// logits = h1 @ out_w^T + out_b; only per-row argmax escapes. 250 blocks x
// 8 waves cover the 2000 N-tiles of V=32000. h1 staged to LDS via TDM.
__global__ __launch_bounds__(256)
void logits_argmax_kernel(const bf16_t* __restrict__ h1,
                          const bf16_t* __restrict__ Wout,
                          const float* __restrict__ outb,
                          unsigned long long* __restrict__ packed) {
  __shared__ bf16_t sH[BB * HH];          // 16 KB
  __shared__ float gbuf[8][16][16];
  __shared__ float bestv[8][16];
  __shared__ int   bestn[8][16];
  const int tid = threadIdx.x, wave = tid >> 5, lane = tid & 31;
  const int ncb = (blockIdx.x * 8 + wave) * 16;
  const int m = lane & 15, khalf = lane >> 4;

  if (tid == 0) {
    tdm_load_to_lds(sH, h1, (unsigned)(BB * HH * sizeof(bf16_t) / 4));
    __builtin_amdgcn_s_wait_tensorcnt(0);
  }
  __builtin_prefetch(Wout + (size_t)(ncb + m) * HH, 0, 0);
  __builtin_prefetch(Wout + (size_t)(ncb + m) * HH + 256, 0, 0);
  __syncthreads();

  v8f acc = {};
  for (int kt = 0; kt < HH; kt += 32) {
    v16bf a = load_frag_a(sH, HH, m, khalf, kt);
    v16bf b = load_frag_b(Wout, HH, ncb, lane, kt);
    acc = __builtin_amdgcn_wmma_f32_16x16x32_bf16(false, a, false, b,
                                                  (short)0, acc, false, false);
  }
  const float bv = outb[ncb + m];
  const int mbase = khalf * 8;
#pragma unroll
  for (int r = 0; r < 8; ++r) gbuf[wave][mbase + r][m] = acc[r] + bv;
  __syncthreads();

  if (lane < 16) {            // lane == batch row m for this wave's tile
    float v0 = gbuf[wave][lane][0];
    int   n0 = ncb;
    for (int j = 1; j < 16; ++j) {
      float v = gbuf[wave][lane][j];
      if (v > v0) { v0 = v; n0 = ncb + j; }   // ties keep lowest index
    }
    bestv[wave][lane] = v0;
    bestn[wave][lane] = n0;
  }
  __syncthreads();

  if (tid < BB) {
    float v0 = bestv[0][tid];
    int   n0 = bestn[0][tid];
    for (int w = 1; w < 8; ++w) {
      float v = bestv[w][tid];
      int   n = bestn[w][tid];
      if (v > v0 || (v == v0 && n < n0)) { v0 = v; n0 = n; }
    }
    unsigned u = __float_as_uint(v0);
    u = (u & 0x80000000u) ? ~u : (u | 0x80000000u);   // orderable float bits
    unsigned long long p =
        ((unsigned long long)u << 32) | (unsigned)(0x7FFFFFFF - n0);
    atomicMax(packed + tid, p);
  }
}

__global__ void decode_last_kernel(const unsigned long long* __restrict__ packed,
                                   int* __restrict__ res) {
  int tid = threadIdx.x;
  if (tid < BB) {
    unsigned long long p = packed[tid];
    res[(SS - 1) * BB + tid] = 0x7FFFFFFF - (int)(unsigned)(p & 0xFFFFFFFFull);
  }
}

// -------- straight-through forward value: out = transfer_w[res] | tail ------
__global__ void output_kernel(const int* __restrict__ res,
                              const float* __restrict__ transfer_w,
                              const float* __restrict__ sentences,
                              float* __restrict__ out) {
  int idx = blockIdx.x * blockDim.x + threadIdx.x;
  const int total = BB * TT * EE;
  if (idx >= total) return;
  const int e = idx % EE;
  const int t = (idx / EE) % TT;
  const int b = idx / (EE * TT);
  if (t < SS) out[idx] = transfer_w[(size_t)res[t * BB + b] * EE + e];
  else        out[idx] = sentences[idx];
  (void)b;
}

// ------------------------------- host driver --------------------------------
extern "C" void kernel_launch(void* const* d_in, const int* in_sizes, int n_in,
                              void* d_out, int out_size, void* d_ws, size_t ws_size,
                              hipStream_t stream) {
  const float* hiddens    = (const float*)d_in[0];
  const float* sentences  = (const float*)d_in[1];
  const int*   relations  = (const int*)d_in[2];
  const int*   gold       = (const int*)d_in[3];
  const int*   tmask      = (const int*)d_in[4];
  const float* embed      = (const float*)d_in[5];
  const float* rel_emb    = (const float*)d_in[6];
  const float* w_ih0      = (const float*)d_in[7];
  const float* w_hh0      = (const float*)d_in[8];
  const float* b0         = (const float*)d_in[9];
  const float* w_ih1      = (const float*)d_in[10];
  const float* w_hh1      = (const float*)d_in[11];
  const float* b1         = (const float*)d_in[12];
  const float* out_w      = (const float*)d_in[13];
  const float* out_b      = (const float*)d_in[14];
  const float* transfer_w = (const float*)d_in[15];

  char* ws = (char*)d_ws;
  size_t off = 0;
  auto take = [&](size_t bytes) -> char* {
    char* p = ws + off;
    off += (bytes + 255) & ~(size_t)255;
    return p;
  };

  bf16_t* outw_bf = (bf16_t*)take((size_t)VV * HH * 2);    // 32.8 MB, lives in L2
  bf16_t* wih0_bf = (bf16_t*)take((size_t)G4H * KIN * 2);
  bf16_t* whh0_bf = (bf16_t*)take((size_t)G4H * HH * 2);
  bf16_t* wih1_bf = (bf16_t*)take((size_t)G4H * HH * 2);
  bf16_t* whh1_bf = (bf16_t*)take((size_t)G4H * HH * 2);
  bf16_t* x_bf    = (bf16_t*)take((size_t)BB * KIN * 2);
  bf16_t* h0bf[2] = {(bf16_t*)take(BB * HH * 2), (bf16_t*)take(BB * HH * 2)};
  bf16_t* h1bf[2] = {(bf16_t*)take(BB * HH * 2), (bf16_t*)take(BB * HH * 2)};
  float*  c0f[2]  = {(float*)take(BB * HH * 4), (float*)take(BB * HH * 4)};
  float*  c1f[2]  = {(float*)take(BB * HH * 4), (float*)take(BB * HH * 4)};
  int* res = (int*)take(SS * BB * 4);
  unsigned long long* packed = (unsigned long long*)take(BB * 8);

  // one-time fp32 -> bf16 weight conversion (amortized over 128 steps)
  f32_to_bf16_kernel<<<2048, 256, 0, stream>>>(out_w, outw_bf, VV * HH);
  f32_to_bf16_kernel<<<1024, 256, 0, stream>>>(w_ih0, wih0_bf, G4H * KIN);
  f32_to_bf16_kernel<<<1024, 256, 0, stream>>>(w_hh0, whh0_bf, G4H * HH);
  f32_to_bf16_kernel<<<1024, 256, 0, stream>>>(w_ih1, wih1_bf, G4H * HH);
  f32_to_bf16_kernel<<<1024, 256, 0, stream>>>(w_hh1, whh1_bf, G4H * HH);
  // initial state written into the "prev" buffers of step 0 (index 1)
  init_state_kernel<<<(BB * HH + 255) / 256, 256, 0, stream>>>(
      hiddens, h0bf[1], h1bf[1], c0f[1], c1f[1]);

  const size_t shm0 = (size_t)BB * (KIN + HH) * 2 + (size_t)BB * G4H * 4;
  const size_t shm1 = (size_t)BB * (HH + HH) * 2 + (size_t)BB * G4H * 4;

  for (int s = 0; s < SS; ++s) {
    const int cur = s & 1, prv = cur ^ 1;
    prep_kernel<<<1, 256, 0, stream>>>(s, relations, gold, tmask, embed,
                                       rel_emb, packed, res, x_bf);
    lstm_layer_kernel<<<1, 1024, shm0, stream>>>(
        x_bf, KIN, wih0_bf, h0bf[prv], HH, whh0_bf, b0,
        c0f[prv], c0f[cur], h0bf[cur]);
    lstm_layer_kernel<<<1, 1024, shm1, stream>>>(
        h0bf[cur], HH, wih1_bf, h1bf[prv], HH, whh1_bf, b1,
        c1f[prv], c1f[cur], h1bf[cur]);
    logits_argmax_kernel<<<VV / 128, 256, 0, stream>>>(
        h1bf[cur], outw_bf, out_b, packed);
  }
  decode_last_kernel<<<1, 32, 0, stream>>>(packed, res);
  output_kernel<<<(BB * TT * EE + 255) / 256, 256, 0, stream>>>(
      res, transfer_w, sentences, (float*)d_out);
  (void)in_sizes; (void)n_in; (void)out_size; (void)ws_size;
}
